// MultiAttn_51522427683456
// MI455X (gfx1250) — matrix operand order repaired and checked by reading, not compiled, not run on hardware
//
#include <hip/hip_runtime.h>
#include <hip/hip_bf16.h>

// MI455X / gfx1250: wave32, WMMA. Memory-bound workload (KV concat = ~807MB
// of traffic @ 23.3 TB/s ~= 35us dominates 8.6 GFLOP of GEMM). bf16 WMMA with
// f32 accumulation for all matmul stages; async Global<->LDS engine for the
// KV-cache copy.

#define DEV __device__ __forceinline__

typedef __attribute__((ext_vector_type(16))) __bf16 v16bf;
typedef __attribute__((ext_vector_type(8)))  float  v8f;
typedef __attribute__((ext_vector_type(4)))  int    v4i;

constexpr int Bn = 8, Sn = 16, Hn = 16, Dh = 128, Dm = 2048, Pn = 4096, PS = 4112;
constexpr float MIN_F16 = -65504.0f;
constexpr float QSCALE  = 0.08838834764831845f; // 1/sqrt(128)

// Native bf16 convert (v_cvt_*_bf16_f32) instead of software RNE bit-twiddling.
DEV __bf16 f2bf(float f) { return (__bf16)f; }

// ds_swizzle XOR butterflies within 16-lane half-wave groups (wave32).
#define SWZ(v, imm) __builtin_bit_cast(float, \
    __builtin_amdgcn_ds_swizzle(__builtin_bit_cast(int, (v)), (imm)))

DEV float rmax16(float v) {
  v = fmaxf(v, SWZ(v, 0x041F));  // xor 1
  v = fmaxf(v, SWZ(v, 0x081F));  // xor 2
  v = fmaxf(v, SWZ(v, 0x101F));  // xor 4
  v = fmaxf(v, SWZ(v, 0x201F));  // xor 8
  return v;
}
DEV float rsum16(float v) {
  v += SWZ(v, 0x041F);
  v += SWZ(v, 0x081F);
  v += SWZ(v, 0x101F);
  v += SWZ(v, 0x201F);
  return v;
}

// A-matrix (16x32 bf16) gather from row-major f32 source.
// Contiguous k order used here: lanes0-15 row=lane, K=0..15; lanes16-31 row=lane-16, K=16..31.
DEV v16bf load_A(const float* src, int ld, int mbase, int kbase, int lane) {
  int ln = lane & 15, hi = (lane >> 4) & 1;
  const float* row = src + (size_t)(mbase + ln) * ld + kbase;
  v16bf a;
#pragma unroll
  for (int h = 0; h < 16; ++h) {
    int j = h >> 1, e = h & 1;
    int k = 2 * j + e + (hi ? 16 : 0);
    a[h] = f2bf(row[k]);
  }
  return a;
}

// B-matrix (32x16 bf16) gather from row-major f32 W[k][n].
// Contiguous k order used here: lanes0-15 hold K=0..15 (2/VGPR), lanes16-31 hold K=16..31; N=lane%16.
DEV v16bf load_Brm(const float* w, int ld, int kbase, int nbase, int lane) {
  int ln = lane & 15, hi = (lane >> 4) & 1;
  v16bf b;
#pragma unroll
  for (int h = 0; h < 16; ++h) {
    int j = h >> 1, e = h & 1;
    int k = 2 * j + e + (hi ? 16 : 0);
    b[h] = f2bf(w[(size_t)(kbase + k) * ld + nbase + ln]);
  }
  return b;
}

// B-fragment of K^T: element (k=d, n=key) = K[key][d]; key-major f32 [PS][128].
DEV v16bf load_kT(const float* ksl, int keybase, int dbase, int lane) {
  int ln = lane & 15, hi = (lane >> 4) & 1;
  int key = keybase + ln; if (key > PS - 1) key = PS - 1;
  const float* row = ksl + (size_t)key * Dh + dbase;
  v16bf b;
#pragma unroll
  for (int h = 0; h < 16; ++h) {
    int j = h >> 1, e = h & 1;
    int k = 2 * j + e + (hi ? 16 : 0);
    b[h] = f2bf(row[k]);
  }
  return b;
}

// B-fragment of V: element (k=key_in_block, n=d) = V[key][d].
DEV v16bf load_V(const float* vsl, int keybase, int dbase, int lane) {
  int ln = lane & 15, hi = (lane >> 4) & 1;
  v16bf b;
#pragma unroll
  for (int h = 0; h < 16; ++h) {
    int j = h >> 1, e = h & 1;
    int kk = keybase + 2 * j + e + (hi ? 16 : 0);
    if (kk > PS - 1) kk = PS - 1;           // epilogue pad (P=0 kills it)
    b[h] = f2bf(vsl[(size_t)kk * Dh + dbase + ln]);
  }
  return b;
}

// ---------------------------------------------------------------------------
// Kernel 1: stream prefix_k/prefix_v into the [2,B,H,PS,Dh] output KV buffer.
// Each (c,b,h) slice is a contiguous 2MB -> contiguous 2MB copy.
// grid = (128 chunks of 16KB, 256 slices), block = 256.
#if __has_builtin(__builtin_amdgcn_global_load_async_to_lds_b128) && \
    __has_builtin(__builtin_amdgcn_global_store_async_from_lds_b128) && \
    __has_builtin(__builtin_amdgcn_s_wait_asynccnt)
#define HAVE_ASYNC_LDS 1
#endif

#define AS1 __attribute__((address_space(1)))
#define AS3 __attribute__((address_space(3)))

__global__ void copy_prefix(const float4* __restrict__ pk,
                            const float4* __restrict__ pv,
                            float4* __restrict__ okv) {
  const int SL4 = Pn * Dh / 4;        // 131072 float4 per (b,h) slice
  const int DL4 = PS * Dh / 4;        // 131584
  int cs = blockIdx.y;                // c*128 + bh
  int bh = cs & 127;
  const float4* src = (cs & 128) ? pv : pk;
  src += (size_t)bh * SL4;
  float4* dst = okv + (size_t)cs * DL4;
  int base = blockIdx.x * 1024 + threadIdx.x;   // 4 iterations of 256 float4
#ifdef HAVE_ASYNC_LDS
  __shared__ float4 stage[256];
  AS3 v4i* lstage = (AS3 v4i*)&stage[threadIdx.x];
#pragma unroll
  for (int it = 0; it < 4; ++it) {
    int idx = base + it * 256;
    __builtin_amdgcn_global_load_async_to_lds_b128(
        (AS1 v4i*)(src + idx), lstage, 0, 0);
    __builtin_amdgcn_s_wait_asynccnt(0);
    __builtin_amdgcn_global_store_async_from_lds_b128(
        (AS1 v4i*)(dst + idx), lstage, 0, 0);
    __builtin_amdgcn_s_wait_asynccnt(0);
  }
#else
#pragma unroll
  for (int it = 0; it < 4; ++it) {
    int idx = base + it * 256;
    const v4i* s = (const v4i*)(src + idx);
    v4i* d = (v4i*)(dst + idx);
    v4i v = __builtin_nontemporal_load(s);
    __builtin_nontemporal_store(v, d);
  }
#endif
}

// ---------------------------------------------------------------------------
// Kernel 2: fused QKV projection. M=128 (B*S), N=6144, K=2048.
// One 16x16 tile per wave; q -> workspace (pre-scaled), k/v -> KV buffer rows P..P+15.
__global__ void qkv_gemm(const float* __restrict__ x, const float* __restrict__ Wp,
                         const float* __restrict__ bp, float* __restrict__ qws,
                         float* __restrict__ okv) {
  int lane = threadIdx.x & 31, wv = threadIdx.x >> 5;
  int tile = blockIdx.x * 8 + wv;
  int mt = tile / 384, nt = tile % 384;
  int ln = lane & 15, hi = (lane >> 4) & 1;
  v8f acc = {};
  for (int k0 = 0; k0 < Dm; k0 += 32) {
    __builtin_prefetch(&Wp[(size_t)(k0 + 32) * (3 * Dm) + nt * 16 + ln], 0, 1);
    v16bf a = load_A(x, Dm, mt * 16, k0, lane);
    v16bf b = load_Brm(Wp, 3 * Dm, k0, nt * 16, lane);
    acc = __builtin_amdgcn_wmma_f32_16x16x32_bf16(false, a, false, b,
                                                  (short)0, acc, false, false);
  }
  int n = nt * 16 + ln;
  float bias = bp[n];
#pragma unroll
  for (int i = 0; i < 8; ++i) {
    int M = mt * 16 + i + hi * 8;
    int bb = M >> 4, ss = M & 15;
    float val = acc[i] + bias;
    if (n < Dm) {                                 // q
      int hh = n >> 7, dd = n & 127;
      qws[(((size_t)bb * Hn + hh) * Sn + ss) * Dh + dd] = val * QSCALE;
    } else if (n < 2 * Dm) {                      // k -> kv[0][b][h][P+ss][d]
      int n2 = n - Dm; int hh = n2 >> 7, dd = n2 & 127;
      okv[((size_t)(bb * Hn + hh)) * PS * Dh + (size_t)(Pn + ss) * Dh + dd] = val;
    } else {                                      // v -> kv[1][b][h][P+ss][d]
      int n2 = n - 2 * Dm; int hh = n2 >> 7, dd = n2 & 127;
      okv[((size_t)(Bn * Hn + bb * Hn + hh)) * PS * Dh + (size_t)(Pn + ss) * Dh + dd] = val;
    }
  }
}

// ---------------------------------------------------------------------------
// Kernel 3: flash-style attention. One wave per (b,h); 129 key-blocks of 32.
__global__ void attention(const float* __restrict__ qws, const float* __restrict__ okv,
                          const int* __restrict__ maskp, float* __restrict__ attnws) {
  __shared__ __bf16 lp[4][16][32];     // per-wave P-tile transpose staging
  int lane = threadIdx.x & 31, w = threadIdx.x >> 5;
  int ln = lane & 15, hi = (lane >> 4) & 1;
  int bh = blockIdx.x * 4 + w;
  int b = bh >> 4, h = bh & 15;

  const float* ksl = okv + (size_t)bh * PS * Dh;
  const float* vsl = okv + (size_t)(Bn * Hn + bh) * PS * Dh;
  const float* qsl = qws + (size_t)bh * Sn * Dh;

  v16bf qa[4];
#pragma unroll
  for (int t = 0; t < 4; ++t) qa[t] = load_A(qsl, Dh, 0, t * 32, lane);

  float m[8], l[8];
  v8f o[8] = {};
#pragma unroll
  for (int i = 0; i < 8; ++i) { m[i] = -3.0e38f; l[i] = 0.0f; }

  for (int kb = 0; kb < 129; ++kb) {
    int keys0 = kb * 32;
    v8f c0 = {}, c1 = {};
#pragma unroll
    for (int t = 0; t < 4; ++t) {
      v16bf b0 = load_kT(ksl, keys0, t * 32, lane);
      c0 = __builtin_amdgcn_wmma_f32_16x16x32_bf16(false, qa[t], false, b0,
                                                   (short)0, c0, false, false);
      v16bf b1 = load_kT(ksl, keys0 + 16, t * 32, lane);
      c1 = __builtin_amdgcn_wmma_f32_16x16x32_bf16(false, qa[t], false, b1,
                                                   (short)0, c1, false, false);
    }
    const int* mbase = maskp + (size_t)b * Sn * PS;
    float sc[8];
#pragma unroll
    for (int i = 0; i < 8; ++i) {
      int srow = i + hi * 8;
      const int* mrow = mbase + (size_t)srow * PS;
      int key0 = keys0 + ln;
      float s0 = c0[i], s1 = c1[i];
      if (mrow[key0] == 0) s0 = MIN_F16;
      if (kb < 128) { if (mrow[key0 + 16] == 0) s1 = MIN_F16; }
      else s1 = MIN_F16;                         // pad keys >= 4112
      float mn = fmaxf(m[i], rmax16(fmaxf(s0, s1)));
      sc[i] = __expf(m[i] - mn);
      float p0 = __expf(s0 - mn);
      float p1 = __expf(s1 - mn);
      l[i] = l[i] * sc[i] + rsum16(p0 + p1);
      m[i] = mn;
      lp[w][srow][ln]      = f2bf(p0);
      lp[w][srow][ln + 16] = f2bf(p1);
    }
    __syncthreads();
    v16bf pa;                                    // C-layout -> A-layout via LDS
#pragma unroll
    for (int hh = 0; hh < 16; ++hh) {
      int j = hh >> 1, e = hh & 1;
      int k = 2 * j + e + (hi ? 16 : 0);
      pa[hh] = lp[w][ln][k];
    }
#pragma unroll
    for (int nt = 0; nt < 8; ++nt) {
      v16bf vb = load_V(vsl, keys0, nt * 16, lane);
      v8f a = o[nt];
#pragma unroll
      for (int i = 0; i < 8; ++i) a[i] *= sc[i];
      o[nt] = __builtin_amdgcn_wmma_f32_16x16x32_bf16(false, pa, false, vb,
                                                      (short)0, a, false, false);
    }
    __syncthreads();
  }

#pragma unroll
  for (int i = 0; i < 8; ++i) {
    float inv = 1.0f / l[i];
    int srow = i + hi * 8;
    float* orow = attnws + (size_t)(b * Sn + srow) * Dm + h * Dh;
#pragma unroll
    for (int nt = 0; nt < 8; ++nt) orow[nt * 16 + ln] = o[nt][i] * inv;
  }
}

// ---------------------------------------------------------------------------
// Kernel 4: output projection. M=128, N=2048, K=2048.
__global__ void ff_gemm(const float* __restrict__ attnws, const float* __restrict__ Wf,
                        const float* __restrict__ bf, float* __restrict__ out) {
  int lane = threadIdx.x & 31, wv = threadIdx.x >> 5;
  int tile = blockIdx.x * 8 + wv;
  int mt = tile / 128, nt = tile % 128;
  int ln = lane & 15, hi = (lane >> 4) & 1;
  v8f acc = {};
  for (int k0 = 0; k0 < Dm; k0 += 32) {
    __builtin_prefetch(&Wf[(size_t)(k0 + 32) * Dm + nt * 16 + ln], 0, 1);
    v16bf a = load_A(attnws, Dm, mt * 16, k0, lane);
    v16bf b = load_Brm(Wf, Dm, k0, nt * 16, lane);
    acc = __builtin_amdgcn_wmma_f32_16x16x32_bf16(false, a, false, b,
                                                  (short)0, acc, false, false);
  }
  int n = nt * 16 + ln;
  float bias = bf[n];
#pragma unroll
  for (int i = 0; i < 8; ++i)
    out[(size_t)(mt * 16 + i + hi * 8) * Dm + n] = acc[i] + bias;
}

// ---------------------------------------------------------------------------
extern "C" void kernel_launch(void* const* d_in, const int* in_sizes, int n_in,
                              void* d_out, int out_size, void* d_ws, size_t ws_size,
                              hipStream_t stream) {
  const float* x  = (const float*)d_in[0];
  const float* pk = (const float*)d_in[1];
  const float* pv = (const float*)d_in[2];
  const int*   mk = (const int*)d_in[3];
  const float* Wp = (const float*)d_in[4];
  const float* bp = (const float*)d_in[5];
  const float* Wf = (const float*)d_in[6];
  const float* bf = (const float*)d_in[7];

  float* out   = (float*)d_out;
  float* okv   = out + (size_t)Bn * Sn * Dm;        // next_prefix_kv region
  float* qws   = (float*)d_ws;                      // [B,H,S,128] f32, q/sqrt(Dh)
  float* attnw = qws + (size_t)Bn * Hn * Sn * Dh;   // [B,S,2048] f32

  dim3 cgrid(128, 256);
  copy_prefix<<<cgrid, 256, 0, stream>>>((const float4*)pk, (const float4*)pv,
                                         (float4*)okv);
  qkv_gemm<<<384, 256, 0, stream>>>(x, Wp, bp, qws, okv);
  attention<<<32, 128, 0, stream>>>(qws, okv, mk, attnw);
  ff_gemm<<<128, 256, 0, stream>>>(attnw, Wf, bf, out);
}
